// TemporalSelfAttention_36344013259299
// MI455X (gfx1250) — compile-verified, matrix operand order checked by reading, not run on hardware
//
#include <hip/hip_runtime.h>
#include <math.h>

#define TDIM 1024
#define DDIM 1024
#define BDIM 4
#define HDIM 16
#define DH   64

typedef __attribute__((ext_vector_type(16))) __bf16 v16bf;
typedef __attribute__((ext_vector_type(8)))  float  v8f;
typedef __attribute__((ext_vector_type(4)))  int    v4i;

union FragU { uint4 u[2]; v16bf v; };

#if defined(__gfx1250__) && __has_builtin(__builtin_amdgcn_global_load_async_to_lds_b128)
#define ASYNC_LDS 1
#else
#define ASYNC_LDS 0
#endif

#if ASYNC_LDS
typedef __attribute__((address_space(1))) v4i* as1_v4i_ptr;
typedef __attribute__((address_space(3))) v4i* as3_v4i_ptr;
__device__ __forceinline__ void async_copy16(const __bf16* g, __bf16* l) {
  __builtin_amdgcn_global_load_async_to_lds_b128(
      (as1_v4i_ptr)(__bf16*)g, (as3_v4i_ptr)l, 0, 0);
}
__device__ __forceinline__ void wait_async() {
#if __has_builtin(__builtin_amdgcn_s_wait_asynccnt)
  __builtin_amdgcn_s_wait_asynccnt(0);
#else
  asm volatile("s_wait_asynccnt 0" ::: "memory");
#endif
}
#endif

// A-style fragment gather for 16x32 bf16 tiles (also used for the NT "B" operand,
// whose per-lane K layout mirrors A). lane<16: row=row0+lane, K in {0..7,16..23};
// lane>=16: row=row0+lane-16, K in {8..15,24..31}. Two aligned 16B chunks.
__device__ __forceinline__ v16bf load_frag(const __bf16* __restrict__ p, int ld,
                                           int row0, int k0, int lane) {
  int r  = row0 + (lane & 15);
  int kb = k0 + ((lane >> 4) << 3);
  const __bf16* base = p + (size_t)r * ld + kb;
  FragU f;
  f.u[0] = *(const uint4*)(base);
  f.u[1] = *(const uint4*)(base + 16);
  return f.v;
}

__device__ __forceinline__ v8f wmma_bf16(v16bf a, v16bf b, v8f c) {
  return __builtin_amdgcn_wmma_f32_16x16x32_bf16(false, a, false, b, (short)0, c,
                                                 false, false);
}

// ---------------- weight convert + transpose (f32 KxN -> bf16 NxK) -------------
__global__ __launch_bounds__(256) void wconv_kernel(
    const float* __restrict__ w0, const float* __restrict__ w1,
    const float* __restrict__ w2, const float* __restrict__ w3,
    __bf16* __restrict__ o0, __bf16* __restrict__ o1,
    __bf16* __restrict__ o2, __bf16* __restrict__ o3) {
  __shared__ float tile[32][33];
  const float* w; __bf16* o;
  switch (blockIdx.z) {
    case 0:  w = w0; o = o0; break;
    case 1:  w = w1; o = o1; break;
    case 2:  w = w2; o = o2; break;
    default: w = w3; o = o3; break;
  }
  int k0 = blockIdx.x * 32, n0 = blockIdx.y * 32;
  int tx = threadIdx.x, ty = threadIdx.y;   // 32 x 8
#pragma unroll
  for (int j = 0; j < 4; ++j)
    tile[ty * 4 + j][tx] = w[(size_t)(k0 + ty * 4 + j) * DDIM + n0 + tx];
  __syncthreads();
#pragma unroll
  for (int j = 0; j < 4; ++j)
    o[(size_t)(n0 + ty * 4 + j) * DDIM + k0 + tx] = (__bf16)tile[tx][ty * 4 + j];
}

// ---------------- block reduction helper (256 threads, 8 waves) ----------------
__device__ __forceinline__ float block_sum_1024(float v, float* red, int tid) {
#pragma unroll
  for (int off = 16; off > 0; off >>= 1) v += __shfl_xor(v, off, 32);
  __syncthreads();
  if ((tid & 31) == 0) red[tid >> 5] = v;
  __syncthreads();
  float s = red[0];
#pragma unroll
  for (int i = 1; i < 8; ++i) s += red[i];
  return s;
}

struct BF4 { __bf16 x, y, z, w; };

// ---------------- LayerNorm(x) -> bf16 -----------------------------------------
__global__ __launch_bounds__(256) void ln_x_kernel(
    const float* __restrict__ x, const float* __restrict__ g,
    const float* __restrict__ bta, __bf16* __restrict__ out) {
  __shared__ float red[8];
  int row = blockIdx.x, tid = threadIdx.x;
  float4 xv = ((const float4*)(x + (size_t)row * DDIM))[tid];
  float mu = block_sum_1024(xv.x + xv.y + xv.z + xv.w, red, tid) * (1.f / 1024.f);
  float dx = xv.x - mu, dy = xv.y - mu, dz = xv.z - mu, dw = xv.w - mu;
  float var = block_sum_1024(dx * dx + dy * dy + dz * dz + dw * dw, red, tid) *
              (1.f / 1024.f);
  float rs = rsqrtf(var + 1e-5f);
  float4 gv = ((const float4*)g)[tid];
  float4 bv = ((const float4*)bta)[tid];
  BF4 o;
  o.x = (__bf16)(dx * rs * gv.x + bv.x);
  o.y = (__bf16)(dy * rs * gv.y + bv.y);
  o.z = (__bf16)(dz * rs * gv.z + bv.z);
  o.w = (__bf16)(dw * rs * gv.w + bv.w);
  *(BF4*)(out + (size_t)row * DDIM + tid * 4) = o;
}

// ---------------- AdaLN embedding MLP: silu(emb) @ emb_W + emb_b ---------------
__global__ __launch_bounds__(256) void emb_kernel(
    const float* __restrict__ emb, const float* __restrict__ W,
    const float* __restrict__ bias, float* __restrict__ out) {
  __shared__ float se[1024];
  int b = blockIdx.y;
  for (int i = threadIdx.x; i < 1024; i += 256) {
    float e = emb[b * 1024 + i];
    se[i] = e / (1.f + __expf(-e));
  }
  __syncthreads();
  int col = blockIdx.x * 256 + threadIdx.x;   // 0..2047
  float acc = bias[col];
  for (int kk = 0; kk < 1024; ++kk)
    acc = fmaf(se[kk], W[(size_t)kk * 2048 + col], acc);
  out[(size_t)b * 2048 + col] = acc;
}

// ---------------- NT WMMA GEMM: C = A(MxK) * Bt(NxK)^T + bias (+resid) ---------
// Double-buffered LDS, async global->LDS copies, last k-step peeled so the main
// loop body is branch-free.
__global__ __launch_bounds__(256) void gemm_bf16_nt(
    const __bf16* __restrict__ A, const __bf16* __restrict__ Bt,
    const float* __restrict__ bias, __bf16* __restrict__ Cbf,
    float* __restrict__ Cf, const float* __restrict__ resid,
    int M, int N, int K) {
  __shared__ __bf16 As[2][128][40];
  __shared__ __bf16 Bs[2][128][40];
  int tid = threadIdx.x, lane = tid & 31, wave = tid >> 5;
  int wm = wave >> 1, wn = wave & 1;          // 4x2 waves; wave tile 32x64
  int m0 = blockIdx.y * 128, n0 = blockIdx.x * 128;
  v8f acc[2][4];
#pragma unroll
  for (int i = 0; i < 2; ++i)
#pragma unroll
    for (int j = 0; j < 4; ++j) acc[i][j] = (v8f){};
  int r = tid >> 2;
  int c = (tid & 3) << 3;
  const __bf16* ga0 = A  + (size_t)(m0 + r) * K + c;
  const __bf16* ga1 = A  + (size_t)(m0 + r + 64) * K + c;
  const __bf16* gb0 = Bt + (size_t)(n0 + r) * K + c;
  const __bf16* gb1 = Bt + (size_t)(n0 + r + 64) * K + c;
  int p = 0;

  auto compute = [&](int pb) {
    v16bf af[2], bfr[4];
#pragma unroll
    for (int mi = 0; mi < 2; ++mi)
      af[mi] = load_frag(&As[pb][0][0], 40, wm * 32 + mi * 16, 0, lane);
#pragma unroll
    for (int ni = 0; ni < 4; ++ni)
      bfr[ni] = load_frag(&Bs[pb][0][0], 40, wn * 64 + ni * 16, 0, lane);
#pragma unroll
    for (int mi = 0; mi < 2; ++mi)
#pragma unroll
      for (int ni = 0; ni < 4; ++ni)
        acc[mi][ni] = wmma_bf16(af[mi], bfr[ni], acc[mi][ni]);
  };

#if ASYNC_LDS
  async_copy16(ga0, &As[0][r][c]);
  async_copy16(ga1, &As[0][r + 64][c]);
  async_copy16(gb0, &Bs[0][r][c]);
  async_copy16(gb1, &Bs[0][r + 64][c]);
  wait_async();
  __syncthreads();
  for (int k0 = 0; k0 + 32 < K; k0 += 32) {   // branch-free body; last step peeled
    async_copy16(ga0 + k0 + 32, &As[p ^ 1][r][c]);
    async_copy16(ga1 + k0 + 32, &As[p ^ 1][r + 64][c]);
    async_copy16(gb0 + k0 + 32, &Bs[p ^ 1][r][c]);
    async_copy16(gb1 + k0 + 32, &Bs[p ^ 1][r + 64][c]);
    compute(p);
    wait_async();
    __syncthreads();
    p ^= 1;
  }
  compute(p);
#else
  uint4 ra0 = *(const uint4*)ga0, ra1 = *(const uint4*)ga1;
  uint4 rb0 = *(const uint4*)gb0, rb1 = *(const uint4*)gb1;
  for (int k0 = 0; k0 + 32 < K; k0 += 32) {
    *(uint4*)&As[p][r][c]      = ra0;
    *(uint4*)&As[p][r + 64][c] = ra1;
    *(uint4*)&Bs[p][r][c]      = rb0;
    *(uint4*)&Bs[p][r + 64][c] = rb1;
    __syncthreads();
    ra0 = *(const uint4*)(ga0 + k0 + 32);
    ra1 = *(const uint4*)(ga1 + k0 + 32);
    rb0 = *(const uint4*)(gb0 + k0 + 32);
    rb1 = *(const uint4*)(gb1 + k0 + 32);
    compute(p);
    p ^= 1;
  }
  *(uint4*)&As[p][r][c]      = ra0;
  *(uint4*)&As[p][r + 64][c] = ra1;
  *(uint4*)&Bs[p][r][c]      = rb0;
  *(uint4*)&Bs[p][r + 64][c] = rb1;
  __syncthreads();
  compute(p);
#endif

#pragma unroll
  for (int mi = 0; mi < 2; ++mi)
#pragma unroll
    for (int ni = 0; ni < 4; ++ni) {
      int nb = n0 + wn * 64 + ni * 16 + (lane & 15);
      int mb = m0 + wm * 32 + mi * 16 + ((lane >> 4) << 3);
      float bb = bias ? bias[nb] : 0.f;
#pragma unroll
      for (int rr = 0; rr < 8; ++rr) {
        size_t idx = (size_t)(mb + rr) * N + nb;
        float val = acc[mi][ni][rr] + bb;
        if (resid) val += resid[idx];
        if (Cf)  Cf[idx]  = val;
        if (Cbf) Cbf[idx] = (__bf16)val;
      }
    }
}

// ---------------- per-batch transpose of v (bf16 1024x1024) --------------------
__global__ __launch_bounds__(256) void vtrans_kernel(const __bf16* __restrict__ v,
                                                     __bf16* __restrict__ vT) {
  __shared__ __bf16 tile[32][33];
  int t0 = blockIdx.x * 32, c0 = blockIdx.y * 32, b = blockIdx.z;
  int tx = threadIdx.x, ty = threadIdx.y;
  const __bf16* src = v  + (size_t)b * TDIM * DDIM;
  __bf16*       dst = vT + (size_t)b * TDIM * DDIM;
#pragma unroll
  for (int j = 0; j < 4; ++j)
    tile[ty * 4 + j][tx] = src[(size_t)(t0 + ty * 4 + j) * DDIM + c0 + tx];
  __syncthreads();
#pragma unroll
  for (int j = 0; j < 4; ++j)
    dst[(size_t)(c0 + ty * 4 + j) * TDIM + t0 + tx] = tile[tx][ty * 4 + j];
}

// ---------------- flash attention: one wave = 16 query rows of one (b,h) -------
// 64-key blocks: 16 WMMAs per iteration; the shuffle reductions, accumulator
// rescale and the P LDS round-trip amortize over twice as many columns.
__global__ __launch_bounds__(256) void attn_kernel(
    const __bf16* __restrict__ q, const __bf16* __restrict__ k,
    const __bf16* __restrict__ vT, const float* __restrict__ mask,
    float* __restrict__ y) {
  __shared__ __bf16 Pl[8][16][72];           // per-wave 16x64 P tile, stride 72
  int bh = blockIdx.x;
  int b = bh >> 4, h = bh & 15;
  int tid = threadIdx.x, lane = tid & 31, wave = tid >> 5;
  int m0 = blockIdx.y * 128 + wave * 16;
  const __bf16* qp = q  + (size_t)b * TDIM * DDIM + h * DH;                // ld=D
  const __bf16* kp = k  + (size_t)b * TDIM * DDIM + h * DH;                // ld=D
  const __bf16* vp = vT + (size_t)b * TDIM * DDIM + (size_t)h * DH * TDIM; // ld=T
  v16bf qf0 = load_frag(qp, DDIM, m0, 0, lane);
  v16bf qf1 = load_frag(qp, DDIM, m0, 32, lane);
  v8f o0 = (v8f){}, o1 = (v8f){}, o2 = (v8f){}, o3 = (v8f){};
  float mr[8], lr[8];
#pragma unroll
  for (int i = 0; i < 8; ++i) { mr[i] = -3.0e38f; lr[i] = 0.f; }
  int rbase = m0 + ((lane >> 4) << 3);
  const float scale = 0.125f;                // 1/sqrt(64)
  const float* mrow = mask + (size_t)b * TDIM * TDIM;
  for (int n0 = 0; n0 < TDIM; n0 += 64) {
    v8f s0 = (v8f){}, s1 = (v8f){}, s2 = (v8f){}, s3 = (v8f){};
    {
      v16bf kf;
      kf = load_frag(kp, DDIM, n0 +  0, 0,  lane); s0 = wmma_bf16(qf0, kf, s0);
      kf = load_frag(kp, DDIM, n0 +  0, 32, lane); s0 = wmma_bf16(qf1, kf, s0);
      kf = load_frag(kp, DDIM, n0 + 16, 0,  lane); s1 = wmma_bf16(qf0, kf, s1);
      kf = load_frag(kp, DDIM, n0 + 16, 32, lane); s1 = wmma_bf16(qf1, kf, s1);
      kf = load_frag(kp, DDIM, n0 + 32, 0,  lane); s2 = wmma_bf16(qf0, kf, s2);
      kf = load_frag(kp, DDIM, n0 + 32, 32, lane); s2 = wmma_bf16(qf1, kf, s2);
      kf = load_frag(kp, DDIM, n0 + 48, 0,  lane); s3 = wmma_bf16(qf0, kf, s3);
      kf = load_frag(kp, DDIM, n0 + 48, 32, lane); s3 = wmma_bf16(qf1, kf, s3);
    }
    // V frags (independent of softmax; issued early so latency hides under VALU)
    v16bf vf00 = load_frag(vp, TDIM, 0,  n0,      lane);
    v16bf vf01 = load_frag(vp, TDIM, 0,  n0 + 32, lane);
    v16bf vf10 = load_frag(vp, TDIM, 16, n0,      lane);
    v16bf vf11 = load_frag(vp, TDIM, 16, n0 + 32, lane);
    v16bf vf20 = load_frag(vp, TDIM, 32, n0,      lane);
    v16bf vf21 = load_frag(vp, TDIM, 32, n0 + 32, lane);
    v16bf vf30 = load_frag(vp, TDIM, 48, n0,      lane);
    v16bf vf31 = load_frag(vp, TDIM, 48, n0 + 32, lane);
    int col0 = n0 + (lane & 15);
#pragma unroll
    for (int rr = 0; rr < 8; ++rr) {
      const float* mp = mrow + (size_t)(rbase + rr) * TDIM;
      float a0 = s0[rr] * scale + (1.f - mp[col0])      * -100000.f;
      float a1 = s1[rr] * scale + (1.f - mp[col0 + 16]) * -100000.f;
      float a2 = s2[rr] * scale + (1.f - mp[col0 + 32]) * -100000.f;
      float a3 = s3[rr] * scale + (1.f - mp[col0 + 48]) * -100000.f;
      float mx = fmaxf(fmaxf(a0, a1), fmaxf(a2, a3));
#pragma unroll
      for (int off = 1; off < 16; off <<= 1) mx = fmaxf(mx, __shfl_xor(mx, off, 32));
      mx = fmaxf(mx, mr[rr]);
      float corr = __expf(mr[rr] - mx);
      float p0 = __expf(a0 - mx);
      float p1 = __expf(a1 - mx);
      float p2 = __expf(a2 - mx);
      float p3 = __expf(a3 - mx);
      float ps = (p0 + p1) + (p2 + p3);
#pragma unroll
      for (int off = 1; off < 16; off <<= 1) ps += __shfl_xor(ps, off, 32);
      lr[rr] = lr[rr] * corr + ps;
      mr[rr] = mx;
      o0[rr] *= corr; o1[rr] *= corr; o2[rr] *= corr; o3[rr] *= corr;
      int prow = rr + ((lane >> 4) << 3);
      Pl[wave][prow][(lane & 15)]      = (__bf16)p0;
      Pl[wave][prow][(lane & 15) + 16] = (__bf16)p1;
      Pl[wave][prow][(lane & 15) + 32] = (__bf16)p2;
      Pl[wave][prow][(lane & 15) + 48] = (__bf16)p3;
    }
    __asm__ volatile("" ::: "memory");       // order ds stores vs ds loads (same wave)
    v16bf pf0 = load_frag(&Pl[wave][0][0], 72, 0, 0,  lane);
    v16bf pf1 = load_frag(&Pl[wave][0][0], 72, 0, 32, lane);
    o0 = wmma_bf16(pf0, vf00, o0); o0 = wmma_bf16(pf1, vf01, o0);
    o1 = wmma_bf16(pf0, vf10, o1); o1 = wmma_bf16(pf1, vf11, o1);
    o2 = wmma_bf16(pf0, vf20, o2); o2 = wmma_bf16(pf1, vf21, o2);
    o3 = wmma_bf16(pf0, vf30, o3); o3 = wmma_bf16(pf1, vf31, o3);
    __asm__ volatile("" ::: "memory");
  }
  float* yp = y + (size_t)b * TDIM * DDIM + h * DH;
#pragma unroll
  for (int rr = 0; rr < 8; ++rr) {
    float inv = 1.f / lr[rr];
    size_t rowoff = (size_t)(rbase + rr) * DDIM + (lane & 15);
    yp[rowoff]      = o0[rr] * inv;
    yp[rowoff + 16] = o1[rr] * inv;
    yp[rowoff + 32] = o2[rr] * inv;
    yp[rowoff + 48] = o3[rr] * inv;
  }
}

// ---------------- LN2 + AdaLN modulation + silu -> bf16 ------------------------
__global__ __launch_bounds__(256) void ln2_kernel(
    const float* __restrict__ y, const float* __restrict__ g,
    const float* __restrict__ bta, const float* __restrict__ embout,
    __bf16* __restrict__ out) {
  __shared__ float red[8];
  int row = blockIdx.x, tid = threadIdx.x;
  int b = row >> 10;
  float4 xv = ((const float4*)(y + (size_t)row * DDIM))[tid];
  float mu = block_sum_1024(xv.x + xv.y + xv.z + xv.w, red, tid) * (1.f / 1024.f);
  float dx = xv.x - mu, dy = xv.y - mu, dz = xv.z - mu, dw = xv.w - mu;
  float var = block_sum_1024(dx * dx + dy * dy + dz * dz + dw * dw, red, tid) *
              (1.f / 1024.f);
  float rs = rsqrtf(var + 1e-5f);
  float4 gv = ((const float4*)g)[tid];
  float4 bv = ((const float4*)bta)[tid];
  float4 sc = ((const float4*)(embout + (size_t)b * 2048))[tid];
  float4 sh = ((const float4*)(embout + (size_t)b * 2048 + 1024))[tid];
  float h0 = (dx * rs * gv.x + bv.x) * (1.f + sc.x) + sh.x;
  float h1 = (dy * rs * gv.y + bv.y) * (1.f + sc.y) + sh.y;
  float h2 = (dz * rs * gv.z + bv.z) * (1.f + sc.z) + sh.z;
  float h3 = (dw * rs * gv.w + bv.w) * (1.f + sc.w) + sh.w;
  BF4 o;
  o.x = (__bf16)(h0 / (1.f + __expf(-h0)));
  o.y = (__bf16)(h1 / (1.f + __expf(-h1)));
  o.z = (__bf16)(h2 / (1.f + __expf(-h2)));
  o.w = (__bf16)(h3 / (1.f + __expf(-h3)));
  *(BF4*)(out + (size_t)row * DDIM + tid * 4) = o;
}

extern "C" void kernel_launch(void* const* d_in, const int* in_sizes, int n_in,
                              void* d_out, int out_size, void* d_ws, size_t ws_size,
                              hipStream_t stream) {
  (void)in_sizes; (void)n_in; (void)out_size; (void)ws_size;
  const float* x    = (const float*)d_in[0];
  const float* emb  = (const float*)d_in[1];
  const float* mask = (const float*)d_in[2];
  const float* ln_g = (const float*)d_in[3];
  const float* ln_b = (const float*)d_in[4];
  const float* Wq   = (const float*)d_in[5];
  const float* bq   = (const float*)d_in[6];
  const float* Wk   = (const float*)d_in[7];
  const float* bk   = (const float*)d_in[8];
  const float* Wv   = (const float*)d_in[9];
  const float* bv   = (const float*)d_in[10];
  const float* embW = (const float*)d_in[11];
  const float* embB = (const float*)d_in[12];
  const float* ln2g = (const float*)d_in[13];
  const float* ln2b = (const float*)d_in[14];
  const float* outW = (const float*)d_in[15];
  const float* outB = (const float*)d_in[16];

  char* ws = (char*)d_ws;
  const size_t MB = 1024 * 1024;
  __bf16* WqT = (__bf16*)(ws + 0 * MB);
  __bf16* WkT = (__bf16*)(ws + 2 * MB);
  __bf16* WvT = (__bf16*)(ws + 4 * MB);
  __bf16* WoT = (__bf16*)(ws + 6 * MB);
  __bf16* xn  = (__bf16*)(ws + 8 * MB);   // reused as hb after QKV GEMMs
  __bf16* qb  = (__bf16*)(ws + 16 * MB);
  __bf16* kb  = (__bf16*)(ws + 24 * MB);
  __bf16* vb  = (__bf16*)(ws + 32 * MB);
  __bf16* vTb = (__bf16*)(ws + 40 * MB);
  float*  yb  = (float*)(ws + 48 * MB);   // 16MB f32
  float*  eo  = (float*)(ws + 64 * MB);   // 32KB

  wconv_kernel<<<dim3(32, 32, 4), dim3(32, 8), 0, stream>>>(
      Wq, Wk, Wv, outW, WqT, WkT, WvT, WoT);
  ln_x_kernel<<<dim3(4096), dim3(256), 0, stream>>>(x, ln_g, ln_b, xn);
  emb_kernel<<<dim3(8, 4), dim3(256), 0, stream>>>(emb, embW, embB, eo);
  gemm_bf16_nt<<<dim3(8, 32), dim3(256), 0, stream>>>(
      xn, WqT, bq, qb, nullptr, nullptr, 4096, 1024, 1024);
  gemm_bf16_nt<<<dim3(8, 32), dim3(256), 0, stream>>>(
      xn, WkT, bk, kb, nullptr, nullptr, 4096, 1024, 1024);
  gemm_bf16_nt<<<dim3(8, 32), dim3(256), 0, stream>>>(
      xn, WvT, bv, vb, nullptr, nullptr, 4096, 1024, 1024);
  vtrans_kernel<<<dim3(32, 32, 4), dim3(32, 8), 0, stream>>>(vb, vTb);
  attn_kernel<<<dim3(64, 8), dim3(256), 0, stream>>>(qb, kb, vTb, mask, yb);
  __bf16* hb = xn;  // xn dead after V GEMM
  ln2_kernel<<<dim3(4096), dim3(256), 0, stream>>>(yb, ln2g, ln2b, eo, hb);
  gemm_bf16_nt<<<dim3(8, 32), dim3(256), 0, stream>>>(
      hb, WoT, outB, nullptr, (float*)d_out, x, 4096, 1024, 1024);
}
